// MRConvLayer_24842090840144
// MI455X (gfx1250) — matrix operand order repaired and checked by reading, not compiled, not run on hardware
//
#include <hip/hip_runtime.h>
#include <hip/hip_bf16.h>

// ---------------------------------------------------------------------------
// MRConv layer for MI455X (gfx1250, wave32, WMMA).
//   Stage A (prep):  transpose x -> xf rows, hi/lo f16 split, sq-norms,
//                    embedding-added f32 copy, W hi/lo split.
//   Stage B (knn):   fused  S = X·X^T  (split-f16 WMMA, f32 accum) + per-row
//                    top-9 smallest  key = ||x_j||^2 - 2<x_i,x_j>  (the
//                    +||x_i||^2 term is a per-row constant -> dropped).
//   Stage C (mrconv):gather neighbors, max-aggregate, WMMA linear, store
//                    transposed [OUT_C, N] with coalesced writes.
// ---------------------------------------------------------------------------

typedef __attribute__((ext_vector_type(16))) _Float16 v16h;
typedef __attribute__((ext_vector_type(8)))  _Float16 v8h;
typedef __attribute__((ext_vector_type(8)))  float    v8f;
typedef __attribute__((ext_vector_type(4)))  float    v4f;

#define NREAL  12100
#define NP     12160           // padded rows: multiple of 32 and 64
#define CDIM   128
#define GRIDW  110
#define KNNK   9
#define WAVES2 4
#define CT     (NP / 16)       // 760 column tiles

// workspace layout (bytes, 256-aligned)
#define XH_OFF  ((size_t)0)
#define XL_OFF  (XH_OFF + (size_t)NP * CDIM * 2)
#define SQN_OFF (XL_OFF + (size_t)NP * CDIM * 2)
#define XE_OFF  (SQN_OFF + (size_t)NP * 4)
#define WH_OFF  (XE_OFF + (size_t)NP * CDIM * 4)
#define WL_OFF  (WH_OFF + (size_t)CDIM * CDIM * 2)
#define NBR_OFF (WL_OFF + (size_t)CDIM * CDIM * 2)

static __device__ __forceinline__ v16h hcat(v8h a, v8h b) {
  return __builtin_shufflevector(a, b, 0,1,2,3,4,5,6,7,8,9,10,11,12,13,14,15);
}
static __device__ __forceinline__ v8f wmma16(v16h a, v16h b, v8f c) {
  return __builtin_amdgcn_wmma_f32_16x16x32_f16(false, a, false, b, (short)0, c,
                                                false, false);
}
// branchless insert into ascending top-K kept fully in registers
static __device__ __forceinline__ void topk_insert(float (&td)[KNNK],
                                                   int (&tj)[KNNK],
                                                   float d, int j) {
  float pd = -3.0e38f; int pj = 0;
#pragma unroll
  for (int p = 0; p < KNNK; ++p) {
    float od = td[p]; int oj = tj[p];
    bool lt = d < od, ltp = d < pd;
    td[p] = lt ? (ltp ? pd : d) : od;
    tj[p] = lt ? (ltp ? pj : j) : oj;
    pd = od; pj = oj;
  }
}

// ---------------------------------------------------------------------------
__global__ __launch_bounds__(256) void prep_kernel(
    const float* __restrict__ x, const float* __restrict__ emb,
    _Float16* __restrict__ xh, _Float16* __restrict__ xl,
    float* __restrict__ sqn, float* __restrict__ xe) {
  __shared__ float tile[CDIM][65];
  __shared__ float psum[64][4];
  const int t  = threadIdx.x;
  const int n0 = blockIdx.x * 64;
#pragma unroll
  for (int it = 0; it < (CDIM * 64) / 256; ++it) {
    int idx = it * 256 + t;
    int c = idx >> 6, nn = idx & 63;
    int n = n0 + nn;
    tile[c][nn] = (n < NREAL) ? x[(size_t)c * NREAL + n] : 0.0f;
  }
  __syncthreads();
  const int nn = t >> 2, q = t & 3;
  const int n  = n0 + nn;
  const int c0 = q * 32;
  const bool real = (n < NREAL);
  int relrow = 0;
  if (real) { int r = n / GRIDW; int cc = n - r * GRIDW; relrow = r - cc + (GRIDW - 1); }
  const float* erow = emb + (size_t)relrow * CDIM + c0;
  float s = 0.0f;
  _Float16* xhp = xh + (size_t)n * CDIM + c0;
  _Float16* xlp = xl + (size_t)n * CDIM + c0;
  float*    xep = xe + (size_t)n * CDIM + c0;
#pragma unroll
  for (int ch = 0; ch < 4; ++ch) {
    v8h hv, lv; v4f ea, eb;
#pragma unroll
    for (int e = 0; e < 8; ++e) {
      float v = real ? tile[c0 + ch * 8 + e][nn] : 0.0f;
      s = fmaf(v, v, s);
      _Float16 h = (_Float16)v;
      hv[e] = h;
      lv[e] = (_Float16)(v - (float)h);
      float ev = real ? (v + erow[ch * 8 + e]) : 0.0f;
      if (e < 4) ea[e] = ev; else eb[e - 4] = ev;
    }
    *(v8h*)(xhp + ch * 8)     = hv;
    *(v8h*)(xlp + ch * 8)     = lv;
    *(v4f*)(xep + ch * 8)     = ea;
    *(v4f*)(xep + ch * 8 + 4) = eb;
  }
  psum[nn][q] = s;
  __syncthreads();
  if (q == 0) {
    float tot = psum[nn][0] + psum[nn][1] + psum[nn][2] + psum[nn][3];
    sqn[n] = real ? tot : 1.0e30f;   // padded columns never selected
  }
}

__global__ void wprep_kernel(const float* __restrict__ W,
                             _Float16* __restrict__ wh,
                             _Float16* __restrict__ wl) {
  int i = blockIdx.x * 256 + threadIdx.x;   // 64 blocks * 256 = 16384
  float v = W[i];
  _Float16 h = (_Float16)v;
  wh[i] = h;
  wl[i] = (_Float16)(v - (float)h);
}

// ---------------------------------------------------------------------------
// Fused gram + top-9. Block = 32 rows, 4 waves stride over column tiles.
// B fragments are double-buffered: loads for kf+1 are issued before the
// WMMAs of kf, so one load clause is always in flight under the matrix ops.
__global__ __launch_bounds__(128) void knn_kernel(
    const _Float16* __restrict__ xh, const _Float16* __restrict__ xl,
    const float* __restrict__ sqn, int* __restrict__ nbr) {
  __shared__ float dtile[WAVES2][32][17];
  __shared__ float md[WAVES2][32][KNNK];
  __shared__ int   mj[WAVES2][32][KNNK];

  const int tid  = threadIdx.x;
  const int wave = tid >> 5, lane = tid & 31;
  const int half = lane >> 4, l16 = lane & 15;
  const int row0 = blockIdx.x * 32;

  // Persistent A fragments (hi/lo) for both 16-row M tiles.
  // A k order used here (contiguous, same as B): lane<16 holds K 0..15, lane>=16 K 16..31.
  v16h Ah[2][4], Al[2][4];
#pragma unroll
  for (int m = 0; m < 2; ++m) {
    const _Float16* rh = xh + (size_t)(row0 + m * 16 + l16) * CDIM;
    const _Float16* rl = xl + (size_t)(row0 + m * 16 + l16) * CDIM;
#pragma unroll
    for (int kf = 0; kf < 4; ++kf) {
      int e0 = kf * 32 + (half ? 16 : 0);
      Ah[m][kf] = hcat(*(const v8h*)(rh + e0), *(const v8h*)(rh + e0 + 8));
      Al[m][kf] = hcat(*(const v8h*)(rl + e0), *(const v8h*)(rl + e0 + 8));
    }
  }

  float td[KNNK]; int tj[KNNK];
#pragma unroll
  for (int q = 0; q < KNNK; ++q) { td[q] = 3.0e38f; tj[q] = 0; }
  const int myrow = row0 + lane;

  // B k order used here (contiguous, same as A): lane<16 holds K 0..15 of column l16, lane>=16 K 16..31.
  const size_t bstride = (size_t)WAVES2 * 16 * CDIM;   // elements per jt step
  const _Float16* bh = xh + (size_t)(wave * 16 + l16) * CDIM;
  const _Float16* bl = xl + (size_t)(wave * 16 + l16) * CDIM;
  const float*   sjp = sqn + wave * 16 + l16;
  int jb = wave * 16;
  const int eb0 = half ? 16 : 0;

  for (int it = 0; it < CT / WAVES2; ++it) {
    // Unconditional next-tile prefetch (one stride past the end still lands
    // inside the workspace; speculative prefetch drops bad translations).
    __builtin_prefetch(bh + bstride, 0, 3);
    __builtin_prefetch(bl + bstride, 0, 3);
    float sj = *sjp;

    v16h Bh[2], Bl[2];
    Bh[0] = hcat(*(const v8h*)(bh + eb0), *(const v8h*)(bh + eb0 + 8));
    Bl[0] = hcat(*(const v8h*)(bl + eb0), *(const v8h*)(bl + eb0 + 8));

    // 4 independent accumulation chains (spacing ~4 between same-chain WMMAs)
    v8f a0a = {0.f,0.f,0.f,0.f,0.f,0.f,0.f,0.f};
    v8f a0b = {0.f,0.f,0.f,0.f,0.f,0.f,0.f,0.f};
    v8f a1a = {0.f,0.f,0.f,0.f,0.f,0.f,0.f,0.f};
    v8f a1b = {0.f,0.f,0.f,0.f,0.f,0.f,0.f,0.f};
#pragma unroll
    for (int kf = 0; kf < 4; ++kf) {
      const int p = kf & 1, np = p ^ 1;
      if (kf < 3) {                       // issue next fragment's loads first
        int e1 = (kf + 1) * 32 + eb0;
        Bh[np] = hcat(*(const v8h*)(bh + e1), *(const v8h*)(bh + e1 + 8));
        Bl[np] = hcat(*(const v8h*)(bl + e1), *(const v8h*)(bl + e1 + 8));
      }
      a0a = wmma16(Ah[0][kf], Bh[p], a0a);
      a1a = wmma16(Ah[1][kf], Bh[p], a1a);
      a0b = wmma16(Al[0][kf], Bh[p], a0b);
      a1b = wmma16(Al[1][kf], Bh[p], a1b);
      a0a = wmma16(Ah[0][kf], Bl[p], a0a);
      a1a = wmma16(Ah[1][kf], Bl[p], a1a);
    }
    // D layout: lane holds col l16, rows v + half*8  -> transpose via LDS so
    // each lane owns one row's 16 candidates.
#pragma unroll
    for (int v = 0; v < 8; ++v) {
      dtile[wave][v + half * 8][l16]      = fmaf(-2.0f, a0a[v] + a0b[v], sj);
      dtile[wave][16 + v + half * 8][l16] = fmaf(-2.0f, a1a[v] + a1b[v], sj);
    }
#pragma unroll
    for (int k = 0; k < 16; ++k) {
      float d = dtile[wave][lane][k];
      int   j = jb + k;
      if (__builtin_expect(d < td[KNNK - 1] && j != myrow, 0))
        topk_insert(td, tj, d, j);
    }
    jb  += WAVES2 * 16;
    bh  += bstride;
    bl  += bstride;
    sjp += WAVES2 * 16;
  }

  // merge the 4 wave-partial top-9s per row
#pragma unroll
  for (int q = 0; q < KNNK; ++q) { md[wave][lane][q] = td[q]; mj[wave][lane][q] = tj[q]; }
  __syncthreads();
  if (wave == 0) {
#pragma unroll
    for (int q = 0; q < KNNK; ++q) { td[q] = 3.0e38f; tj[q] = 0; }
    for (int w = 0; w < WAVES2; ++w)
#pragma unroll
      for (int q = 0; q < KNNK; ++q) {
        float d = md[w][lane][q];
        if (d < td[KNNK - 1]) topk_insert(td, tj, d, mj[w][lane][q]);
      }
#pragma unroll
    for (int q = 0; q < KNNK; ++q) nbr[(size_t)myrow * 16 + q] = tj[q];
  }
}

// ---------------------------------------------------------------------------
// Gather + max-aggregate + linear (WMMA), transposed store.
__global__ __launch_bounds__(256) void mrconv_kernel(
    const float* __restrict__ xe, const int* __restrict__ nbr,
    const _Float16* __restrict__ wh, const _Float16* __restrict__ wl,
    float* __restrict__ out) {
  __shared__ float aggr[16][CDIM + 4];
  __shared__ float outt[CDIM][17];
  const int tid = threadIdx.x;
  const int i0  = blockIdx.x * 16;

  { // step 1: aggr[r][c] = max(0, max_k (xe[nbr[k]][c] - xe[r][c]))
    int r = tid >> 4, c0 = (tid & 15) * 8;
    int row = i0 + r;
    const float* xr = xe + (size_t)row * CDIM + c0;
    float self[8], m[8];
    *(v4f*)&self[0] = *(const v4f*)(xr);
    *(v4f*)&self[4] = *(const v4f*)(xr + 4);
#pragma unroll
    for (int e = 0; e < 8; ++e) m[e] = 0.0f;   // self-loop message is 0
#pragma unroll
    for (int k = 0; k < KNNK; ++k) {
      int jn = nbr[(size_t)row * 16 + k];
      const float* xj = xe + (size_t)jn * CDIM + c0;
      float nb[8];
      *(v4f*)&nb[0] = *(const v4f*)(xj);
      *(v4f*)&nb[4] = *(const v4f*)(xj + 4);
#pragma unroll
      for (int e = 0; e < 8; ++e) m[e] = fmaxf(m[e], nb[e] - self[e]);
    }
#pragma unroll
    for (int e = 0; e < 8; ++e) aggr[r][c0 + e] = m[e];
  }
  __syncthreads();

  // step 2: out-tile per wave: [16 rows i] x [16 cols oc], split-f16 WMMA
  const int wave = tid >> 5, lane = tid & 31;
  const int half = lane >> 4, l16 = lane & 15;
  const int oc = wave * 16 + l16;
  const _Float16* wrh = wh + (size_t)oc * CDIM;
  const _Float16* wrl = wl + (size_t)oc * CDIM;

  // Preload all W fragments first (one clause + one wait)...
  v16h Bh[4], Bl[4];
#pragma unroll
  for (int kf = 0; kf < 4; ++kf) {
    int e0 = kf * 32 + (half ? 16 : 0);
    Bh[kf] = hcat(*(const v8h*)(wrh + e0), *(const v8h*)(wrh + e0 + 8));
    Bl[kf] = hcat(*(const v8h*)(wrl + e0), *(const v8h*)(wrl + e0 + 8));
  }
  // ...then build A from LDS with on-the-fly hi/lo split.
  v16h Ah[4], Al[4];
  {
    const float* ar = aggr[l16];
#pragma unroll
    for (int kf = 0; kf < 4; ++kf) {
      int k0 = kf * 32 + half * 16;
      v16h h, l;
#pragma unroll
      for (int e = 0; e < 16; ++e) {
        int k = k0 + e;
        float v = ar[k];
        _Float16 hh = (_Float16)v;
        h[e] = hh;
        l[e] = (_Float16)(v - (float)hh);
      }
      Ah[kf] = h; Al[kf] = l;
    }
  }
  v8f acc = {0.f,0.f,0.f,0.f,0.f,0.f,0.f,0.f};
#pragma unroll
  for (int kf = 0; kf < 4; ++kf) {
    acc = wmma16(Ah[kf], Bh[kf], acc);
    acc = wmma16(Al[kf], Bh[kf], acc);
    acc = wmma16(Ah[kf], Bl[kf], acc);
  }
#pragma unroll
  for (int v = 0; v < 8; ++v) outt[oc][v + half * 8] = acc[v];
  __syncthreads();

  { // step 3: coalesced transposed store  out[oc][i]
    int ocx = tid >> 1, h2 = tid & 1;
#pragma unroll
    for (int e = 0; e < 8; ++e) {
      int i = i0 + h2 * 8 + e;
      if (i < NREAL) out[(size_t)ocx * NREAL + i] = outt[ocx][h2 * 8 + e];
    }
  }
}

// ---------------------------------------------------------------------------
extern "C" void kernel_launch(void* const* d_in, const int* in_sizes, int n_in,
                              void* d_out, int out_size, void* d_ws, size_t ws_size,
                              hipStream_t stream) {
  (void)in_sizes; (void)n_in; (void)out_size; (void)ws_size;
  const float* x   = (const float*)d_in[0];   // [1,128,12100]
  const float* emb = (const float*)d_in[1];   // [289,128]
  const float* W   = (const float*)d_in[2];   // [128,128]
  char* ws = (char*)d_ws;
  _Float16* xh  = (_Float16*)(ws + XH_OFF);
  _Float16* xl  = (_Float16*)(ws + XL_OFF);
  float*    sqn = (float*)(ws + SQN_OFF);
  float*    xe  = (float*)(ws + XE_OFF);
  _Float16* wh  = (_Float16*)(ws + WH_OFF);
  _Float16* wl  = (_Float16*)(ws + WL_OFF);
  int*      nbr = (int*)(ws + NBR_OFF);
  float*    out = (float*)d_out;

  prep_kernel  <<<NP / 64, 256, 0, stream>>>(x, emb, xh, xl, sqn, xe);
  wprep_kernel <<<(CDIM * CDIM) / 256, 256, 0, stream>>>(W, wh, wl);
  knn_kernel   <<<NP / 32, 128, 0, stream>>>(xh, xl, sqn, nbr);
  mrconv_kernel<<<(NREAL + 15) / 16, 256, 0, stream>>>(xe, nbr, wh, wl, out);
}
